// MyGraphConv_64656437674423
// MI455X (gfx1250) — matrix operand order repaired and checked by reading, not compiled, not run on hardware
//
#include <hip/hip_runtime.h>
#include <hip/hip_bf16.h>

#define IN_FEATS  256
#define OUT_FEATS 128

typedef __bf16 v16bf __attribute__((ext_vector_type(16)));
typedef float  v8f   __attribute__((ext_vector_type(8)));

// ---------------------------------------------------------------------------
// Kernel 1: h[n, :] = (feat[n, :] @ W) * cj[n]   via v_wmma_f32_16x16x32_bf16
// One wave -> 16 rows x 128 cols. Block = 256 threads = 8 waves = 128 rows.
// W is converted fp32->bf16 once per block into LDS (64 KB of 320 KB/WGP).
// ---------------------------------------------------------------------------
__global__ __launch_bounds__(256) void gc_gemm_bf16(
    const float* __restrict__ feat, const float* __restrict__ weight,
    const float* __restrict__ cj, float* __restrict__ h, int nNodes)
{
    __shared__ __align__(32) __bf16 wlds[IN_FEATS * OUT_FEATS]; // 64 KB

    // Cooperative fp32 -> bf16 conversion of W into LDS (packed 32-bit stores)
    {
        const float2* wsrc = (const float2*)weight;
        unsigned int* wdst = (unsigned int*)wlds;
        for (int i = threadIdx.x; i < IN_FEATS * OUT_FEATS / 2; i += blockDim.x) {
            float2 v = wsrc[i];
            unsigned short lo = __builtin_bit_cast(unsigned short, (__bf16)v.x);
            unsigned short hi = __builtin_bit_cast(unsigned short, (__bf16)v.y);
            wdst[i] = (unsigned int)lo | ((unsigned int)hi << 16);
        }
    }
    __syncthreads();

    const int lane   = threadIdx.x & 31;
    const int strips = (nNodes + 15) >> 4;
    const int strip  = blockIdx.x * 8 + (threadIdx.x >> 5);
    if (strip >= strips) return;          // whole-wave exit: EXEC stays all-1s

    const int row0 = strip << 4;
    const int hsel = lane >> 4;           // lane group 0/1
    const int m    = lane & 15;
    // wave-uniform: does this strip cover 16 in-range rows?
    const bool fullTile = (row0 + 16 <= nNodes);

    // ---- A: full K=256 strip, per ISA 16-bit A layout ----
    // lane (16*hsel + m) holds row m; tile t: K = 32t + 8*hsel + {0..7}, +16+{0..7}
    int rowA = row0 + m;  if (rowA >= nNodes) rowA = nNodes - 1;
    const float4* ap = (const float4*)(feat + (size_t)rowA * IN_FEATS);
    v16bf a[8];
#pragma unroll
    for (int t = 0; t < 8; ++t) {
        const int b0 = t * 8 + hsel * 2;          // float4 index of K = 32t + 8*hsel
        float4 q0 = ap[b0 + 0];
        float4 q1 = ap[b0 + 1];
        float4 q2 = ap[b0 + 4];                   // K + 16
        float4 q3 = ap[b0 + 5];
        float f[16] = {q0.x,q0.y,q0.z,q0.w, q1.x,q1.y,q1.z,q1.w,
                       q2.x,q2.y,q2.z,q2.w, q3.x,q3.y,q3.z,q3.w};
#pragma unroll
        for (int i = 0; i < 16; ++i) a[t][i] = (__bf16)f[i];
    }

    // cj for the 8 rows this lane will store (C layout: VGPR r -> M = r + 8*hsel)
    float cjv[8];
    if (fullTile) {
#pragma unroll
        for (int r = 0; r < 8; ++r) cjv[r] = cj[row0 + hsel * 8 + r];
    } else {
#pragma unroll
        for (int r = 0; r < 8; ++r) {
            int row = row0 + hsel * 8 + r;
            cjv[r] = cj[row < nNodes ? row : 0];
        }
    }

    // ---- per 16-col output tile: 8 WMMAs over K, then fused cj-scale store ----
#pragma unroll
    for (int j = 0; j < 8; ++j) {
        v8f acc = {0.f, 0.f, 0.f, 0.f, 0.f, 0.f, 0.f, 0.f};
#pragma unroll
        for (int t = 0; t < 8; ++t) {
            // B layout (16-bit, 32x16): lane -> N = m, elements K = 8*hsel + {0..7}, +16+{0..7}
            v16bf b;
#pragma unroll
            for (int s = 0; s < 16; ++s) {
                const int kk = t * 32 + hsel * 8 + (s < 8 ? s : s + 8);
                b[s] = wlds[(size_t)kk * OUT_FEATS + j * 16 + m];
            }
            acc = __builtin_amdgcn_wmma_f32_16x16x32_bf16(
                false, a[t], false, b, (short)0, acc, false, false);
        }
        float* hp = h + (size_t)(row0 + hsel * 8) * OUT_FEATS + j * 16 + m;
        if (fullTile) {
            // wave-uniform fast path: unguarded coalesced stores
#pragma unroll
            for (int r = 0; r < 8; ++r)
                hp[(size_t)r * OUT_FEATS] = acc[r] * cjv[r];
        } else {
#pragma unroll
            for (int r = 0; r < 8; ++r) {
                int row = row0 + hsel * 8 + r;
                if (row < nNodes)
                    h[(size_t)row * OUT_FEATS + j * 16 + m] = acc[r] * cjv[r];
            }
        }
    }
}

// ---------------------------------------------------------------------------
// Kernel 2: zero the output (atomic accumulation base). float4 stores.
// ---------------------------------------------------------------------------
__global__ void gc_zero(float4* __restrict__ out, long n4)
{
    const float4 z = {0.f, 0.f, 0.f, 0.f};
    long stride = (long)gridDim.x * blockDim.x;
    for (long i = (long)blockIdx.x * blockDim.x + threadIdx.x; i < n4; i += stride)
        out[i] = z;
}

// ---------------------------------------------------------------------------
// Kernel 3: edge scatter-sum. One wave per edge (32 lanes x float4 = 128 f32).
// h and out both fit in the 192 MB L2 -> random gather + f32 atomics stay in L2.
// src/dst loads are wave-uniform -> scalarized by the compiler.
// ---------------------------------------------------------------------------
__global__ __launch_bounds__(256) void gc_scatter(
    const float* __restrict__ h, const int* __restrict__ src,
    const int* __restrict__ dst, float* __restrict__ out, int nEdges)
{
    const int lane = threadIdx.x & 31;
    const int nw   = (gridDim.x * blockDim.x) >> 5;
    for (int e = (blockIdx.x * blockDim.x + threadIdx.x) >> 5; e < nEdges; e += nw) {
        const int s = src[e];
        const int d = dst[e];
        const float4 v = ((const float4*)(h + (size_t)s * OUT_FEATS))[lane];
        float* op = out + (size_t)d * OUT_FEATS + lane * 4;
        atomicAdd(op + 0, v.x);
        atomicAdd(op + 1, v.y);
        atomicAdd(op + 2, v.z);
        atomicAdd(op + 3, v.w);
    }
}

// ---------------------------------------------------------------------------
// Kernel 4: out[n, :] *= ci[n]
// ---------------------------------------------------------------------------
__global__ void gc_scale_ci(float4* __restrict__ out, const float* __restrict__ ci,
                            int nNodes)
{
    const int total  = nNodes * (OUT_FEATS / 4);
    const int stride = gridDim.x * blockDim.x;
    for (int i = blockIdx.x * blockDim.x + threadIdx.x; i < total; i += stride) {
        const float c = ci[i >> 5];               // 32 float4 per row
        float4 v = out[i];
        v.x *= c; v.y *= c; v.z *= c; v.w *= c;
        out[i] = v;
    }
}

extern "C" void kernel_launch(void* const* d_in, const int* in_sizes, int n_in,
                              void* d_out, int out_size, void* d_ws, size_t ws_size,
                              hipStream_t stream)
{
    const float* feat   = (const float*)d_in[0];
    const float* weight = (const float*)d_in[1];
    const float* cj     = (const float*)d_in[2];
    const float* ci     = (const float*)d_in[3];
    const int*   src    = (const int*)d_in[4];
    const int*   dst    = (const int*)d_in[5];
    float* out = (float*)d_out;
    float* h   = (float*)d_ws;                    // 100000*128*4 = 51.2 MB scratch

    const int nNodes = in_sizes[2];               // cj has one scalar per node
    const int nEdges = in_sizes[4];

    const int strips     = (nNodes + 15) / 16;
    const int gemmBlocks = (strips + 7) / 8;

    gc_gemm_bf16<<<gemmBlocks, 256, 0, stream>>>(feat, weight, cj, h, nNodes);

    const long n4 = (long)nNodes * (OUT_FEATS / 4);
    gc_zero<<<2048, 256, 0, stream>>>((float4*)out, n4);

    gc_scatter<<<8192, 256, 0, stream>>>(h, src, dst, out, nEdges);

    gc_scale_ci<<<2048, 256, 0, stream>>>((float4*)out, ci, nNodes);
}
